// Attention_55671366091412
// MI455X (gfx1250) — matrix operand order repaired and checked by reading, not compiled, not run on hardware
//
#include <hip/hip_runtime.h>
#include <hip/hip_bf16.h>

// ---------------------------------------------------------------------------
// Types
// ---------------------------------------------------------------------------
typedef __bf16 bf16;
typedef __attribute__((ext_vector_type(16))) __bf16 v16bf;
typedef __attribute__((ext_vector_type(8)))  __bf16 v8bf;
typedef __attribute__((ext_vector_type(8)))  float  v8f;
typedef int v4i __attribute__((vector_size(4 * sizeof(int))));  // matches builtin proto

#define EPS_LN (1.0f / 65530.0f)

#define AS_GLOBAL __attribute__((address_space(1)))
#define AS_LDS    __attribute__((address_space(3)))

#if __has_builtin(__builtin_amdgcn_global_load_async_to_lds_b128) && \
    __has_builtin(__builtin_amdgcn_s_wait_asynccnt)
#define HAVE_ASYNC_LDS 1
#else
#define HAVE_ASYNC_LDS 0
#endif

// float -> bf16, round-to-nearest-even (bit ops; no native-cast reliance)
__device__ __forceinline__ bf16 f2bf(float f) {
  unsigned u = __builtin_bit_cast(unsigned, f);
  unsigned r = u + 0x7FFFu + ((u >> 16) & 1u);
  unsigned short hv = (unsigned short)(r >> 16);
  return __builtin_bit_cast(bf16, hv);
}

// ---------------------------------------------------------------------------
// WMMA fragment loaders (CDNA5 16x16x32 bf16 layouts, ISA 7.12.2)
// A (16x32): lane l -> row m=l%16; elems 0..7 = k (l/16)*8+0..7,
//            elems 8..15 = 16+(l/16)*8+0..7   (two contiguous 16B chunks)
// B (32x16): lane l -> col n=l%16; same k order as A (two 16B chunks),
//            loaded from "B^T in memory" (row n, contiguous k).
// ---------------------------------------------------------------------------
__device__ __forceinline__ v16bf load_a_frag(const bf16* __restrict__ base,
                                             int ld, int lane) {
  int m  = lane & 15;
  int hk = (lane >> 4) << 3;
  const bf16* p = base + (size_t)m * ld + hk;
  v8bf lo = *(const v8bf*)(p);
  v8bf hi = *(const v8bf*)(p + 16);
  v16bf r;
#pragma unroll
  for (int i = 0; i < 8; ++i) { r[i] = lo[i]; r[i + 8] = hi[i]; }
  return r;
}

__device__ __forceinline__ v16bf load_b_frag(const bf16* __restrict__ base,
                                             int ld, int lane) {
  int n  = lane & 15;
  int hk = (lane >> 4) << 3;
  const bf16* p = base + (size_t)n * ld + hk;
  v8bf lo = *(const v8bf*)(p);
  v8bf hi = *(const v8bf*)(p + 16);
  v16bf r;
#pragma unroll
  for (int i = 0; i < 8; ++i) { r[i] = lo[i]; r[i + 8] = hi[i]; }
  return r;
}

// ---------------------------------------------------------------------------
// Elementwise f32 -> bf16 cast (4 elems/thread)
// ---------------------------------------------------------------------------
__global__ __launch_bounds__(256)
void cast_f32_bf16(const float* __restrict__ src, bf16* __restrict__ dst, int n) {
  int i = (blockIdx.x * blockDim.x + threadIdx.x) * 4;
#pragma unroll
  for (int j = 0; j < 4; ++j)
    if (i + j < n) dst[i + j] = f2bf(src[i + j]);
}

// ---------------------------------------------------------------------------
// GEMM: C(MxN,f32) = A(MxK,bf16) @ W(NxK,bf16)^T + bias(N)
// One wave computes a 32x64 tile (2 M-subtiles x 4 N-subtiles): A reused 4x,
// B reused 2x. Two-stage ping-pong pipeline over K (step 64): stage-X loads
// are issued before stage-Y WMMAs with *named* register sets, so no
// inter-stage register copies are required and waits cover loads issued a
// full compute phase earlier. K must be a multiple of 64.
// ---------------------------------------------------------------------------
__global__ __launch_bounds__(256)
void gemm_bf16_kernel(const bf16* __restrict__ A, const bf16* __restrict__ W,
                      const float* __restrict__ bias,
                      float* __restrict__ Cf,
                      int M, int N, int K) {
  int lane = threadIdx.x & 31;
  int wave = blockIdx.x * (blockDim.x >> 5) + (threadIdx.x >> 5);
  int tilesN = N >> 6;
  int tm = wave / tilesN;
  int tn = wave - tm * tilesN;
  if (tm >= (M >> 5)) return;
  int m0 = tm << 5, n0 = tn << 6;
  int col = lane & 15;
  int h   = lane >> 4;

  v8f acc[2][4];
#pragma unroll
  for (int f = 0; f < 4; ++f) {
    float bv = bias ? bias[n0 + f * 16 + col] : 0.0f;
#pragma unroll
    for (int g = 0; g < 2; ++g)
#pragma unroll
      for (int i = 0; i < 8; ++i) acc[g][f][i] = bv;
  }

  const bf16* A0 = A + (size_t)m0 * K;
  const bf16* A1 = A + (size_t)(m0 + 16) * K;
  const bf16* W0 = W + (size_t)(n0 +  0) * K;
  const bf16* W1 = W + (size_t)(n0 + 16) * K;
  const bf16* W2 = W + (size_t)(n0 + 32) * K;
  const bf16* W3 = W + (size_t)(n0 + 48) * K;

  // Stage X: covers k0.  Stage Y: covers k0+32.
  v16bf aX0, aX1, bX0, bX1, bX2, bX3;
  v16bf aY0, aY1, bY0, bY1, bY2, bY3;

  aX0 = load_a_frag(A0, K, lane);
  aX1 = load_a_frag(A1, K, lane);
  bX0 = load_b_frag(W0, K, lane);
  bX1 = load_b_frag(W1, K, lane);
  bX2 = load_b_frag(W2, K, lane);
  bX3 = load_b_frag(W3, K, lane);

  for (int k0 = 0; k0 < K; k0 += 64) {
    int kb = k0 + 32;                       // always valid (K % 64 == 0)
    aY0 = load_a_frag(A0 + kb, K, lane);
    aY1 = load_a_frag(A1 + kb, K, lane);
    bY0 = load_b_frag(W0 + kb, K, lane);
    bY1 = load_b_frag(W1 + kb, K, lane);
    bY2 = load_b_frag(W2 + kb, K, lane);
    bY3 = load_b_frag(W3 + kb, K, lane);
    __builtin_prefetch(A0 + kb + 32, 0, 1);  // -> global_prefetch_b8

    acc[0][0] = __builtin_amdgcn_wmma_f32_16x16x32_bf16(false, aX0, false, bX0, (short)0, acc[0][0], false, false);
    acc[0][1] = __builtin_amdgcn_wmma_f32_16x16x32_bf16(false, aX0, false, bX1, (short)0, acc[0][1], false, false);
    acc[0][2] = __builtin_amdgcn_wmma_f32_16x16x32_bf16(false, aX0, false, bX2, (short)0, acc[0][2], false, false);
    acc[0][3] = __builtin_amdgcn_wmma_f32_16x16x32_bf16(false, aX0, false, bX3, (short)0, acc[0][3], false, false);
    acc[1][0] = __builtin_amdgcn_wmma_f32_16x16x32_bf16(false, aX1, false, bX0, (short)0, acc[1][0], false, false);
    acc[1][1] = __builtin_amdgcn_wmma_f32_16x16x32_bf16(false, aX1, false, bX1, (short)0, acc[1][1], false, false);
    acc[1][2] = __builtin_amdgcn_wmma_f32_16x16x32_bf16(false, aX1, false, bX2, (short)0, acc[1][2], false, false);
    acc[1][3] = __builtin_amdgcn_wmma_f32_16x16x32_bf16(false, aX1, false, bX3, (short)0, acc[1][3], false, false);

    // Reload stage X for k0+64; on the final iteration clamp the address to 0
    // (unconditional dead load, values never consumed) to avoid a swap/branch.
    int ka = (k0 + 64 < K) ? (k0 + 64) : 0;
    aX0 = load_a_frag(A0 + ka, K, lane);
    aX1 = load_a_frag(A1 + ka, K, lane);
    bX0 = load_b_frag(W0 + ka, K, lane);
    bX1 = load_b_frag(W1 + ka, K, lane);
    bX2 = load_b_frag(W2 + ka, K, lane);
    bX3 = load_b_frag(W3 + ka, K, lane);

    acc[0][0] = __builtin_amdgcn_wmma_f32_16x16x32_bf16(false, aY0, false, bY0, (short)0, acc[0][0], false, false);
    acc[0][1] = __builtin_amdgcn_wmma_f32_16x16x32_bf16(false, aY0, false, bY1, (short)0, acc[0][1], false, false);
    acc[0][2] = __builtin_amdgcn_wmma_f32_16x16x32_bf16(false, aY0, false, bY2, (short)0, acc[0][2], false, false);
    acc[0][3] = __builtin_amdgcn_wmma_f32_16x16x32_bf16(false, aY0, false, bY3, (short)0, acc[0][3], false, false);
    acc[1][0] = __builtin_amdgcn_wmma_f32_16x16x32_bf16(false, aY1, false, bY0, (short)0, acc[1][0], false, false);
    acc[1][1] = __builtin_amdgcn_wmma_f32_16x16x32_bf16(false, aY1, false, bY1, (short)0, acc[1][1], false, false);
    acc[1][2] = __builtin_amdgcn_wmma_f32_16x16x32_bf16(false, aY1, false, bY2, (short)0, acc[1][2], false, false);
    acc[1][3] = __builtin_amdgcn_wmma_f32_16x16x32_bf16(false, aY1, false, bY3, (short)0, acc[1][3], false, false);
  }

#pragma unroll
  for (int g = 0; g < 2; ++g)
#pragma unroll
    for (int f = 0; f < 4; ++f)
#pragma unroll
      for (int r = 0; r < 8; ++r) {
        int row = m0 + g * 16 + r + 8 * h;
        Cf[(size_t)row * N + n0 + f * 16 + col] = acc[g][f][r];
      }
}

// ---------------------------------------------------------------------------
// Wave-wide reduction helper (wave32)
// ---------------------------------------------------------------------------
__device__ __forceinline__ float wave_sum32(float v) {
#pragma unroll
  for (int off = 1; off < 32; off <<= 1) v += __shfl_xor(v, off, 32);
  return v;
}

// ---------------------------------------------------------------------------
// LayerNorm + RoPE(half) fused. One wave per (token m, head n);
// lane l owns the RoPE pair (d=l, d=l+32).
// Outputs: Q bf16 [head][s][d] (pre-scaled by 1/sqrt(64)),
//          K bf16 [head][s][d],  V^T bf16 [head][d][s].
// ---------------------------------------------------------------------------
__global__ __launch_bounds__(256)
void ln_rope_kernel(const float* __restrict__ qf,  const float* __restrict__ kvf,
                    const float* __restrict__ cosb, const float* __restrict__ sinb,
                    const float* __restrict__ qnw, const float* __restrict__ qnb,
                    const float* __restrict__ knw, const float* __restrict__ knb,
                    bf16* __restrict__ qo, bf16* __restrict__ ko,
                    bf16* __restrict__ vT) {
  const int S = 2048;
  int lane = threadIdx.x & 31;
  int wave = blockIdx.x * (blockDim.x >> 5) + (threadIdx.x >> 5);
  int n = wave & 15;
  int m = wave >> 4;      // b*S + s
  int b = m >> 11;
  int s = m & 2047;
  int head = b * 16 + n;

  float c1  = cosb[s * 64 + lane], c2  = cosb[s * 64 + lane + 32];
  float sn1 = sinb[s * 64 + lane], sn2 = sinb[s * 64 + lane + 32];

  {  // Q: LN + RoPE + 1/sqrt(D) prescale
    const float* x = qf + (size_t)m * 1024 + n * 64;
    float x1 = x[lane], x2 = x[lane + 32];
    float mean = wave_sum32(x1 + x2) * (1.0f / 64.0f);
    float d1 = x1 - mean, d2 = x2 - mean;
    float var = wave_sum32(d1 * d1 + d2 * d2) * (1.0f / 64.0f);
    float rstd = rsqrtf(var + EPS_LN);
    float y1 = d1 * rstd * qnw[lane]      + qnb[lane];
    float y2 = d2 * rstd * qnw[lane + 32] + qnb[lane + 32];
    bf16* dst = qo + ((size_t)head * S + s) * 64;
    dst[lane]      = f2bf((y1 * c1 - y2 * sn1) * 0.125f);
    dst[lane + 32] = f2bf((y2 * c2 + y1 * sn2) * 0.125f);
  }
  {  // K: LN + RoPE
    const float* x = kvf + (size_t)m * 2048 + n * 64;
    float x1 = x[lane], x2 = x[lane + 32];
    float mean = wave_sum32(x1 + x2) * (1.0f / 64.0f);
    float d1 = x1 - mean, d2 = x2 - mean;
    float var = wave_sum32(d1 * d1 + d2 * d2) * (1.0f / 64.0f);
    float rstd = rsqrtf(var + EPS_LN);
    float y1 = d1 * rstd * knw[lane]      + knb[lane];
    float y2 = d2 * rstd * knw[lane + 32] + knb[lane + 32];
    bf16* dst = ko + ((size_t)head * S + s) * 64;
    dst[lane]      = f2bf(y1 * c1 - y2 * sn1);
    dst[lane + 32] = f2bf(y2 * c2 + y1 * sn2);
  }
  {  // V: transpose-store
    const float* x = kvf + (size_t)m * 2048 + 1024 + n * 64;
    bf16* dst = vT + (size_t)head * 64 * S;
    dst[(size_t)(lane)      * S + s] = f2bf(x[lane]);
    dst[(size_t)(lane + 32) * S + s] = f2bf(x[lane + 32]);
  }
}

// ---------------------------------------------------------------------------
// Flash attention: grid (S/64, B*N), 128 threads = 4 waves.
// Each wave owns 16 query rows. K/V tiles (32 keys) are block-shared, staged
// in LDS via async-to-LDS copies (double-buffered, s_wait_asynccnt) when the
// toolchain exposes the builtins; otherwise plain LDS stores + barrier.
// ---------------------------------------------------------------------------
__global__ __launch_bounds__(128)
void flash_attn_kernel(const bf16* __restrict__ Q, const bf16* __restrict__ Kd,
                       const bf16* __restrict__ Vt, bf16* __restrict__ Aout) {
  const int S = 2048, Dh = 64, Hd = 1024;
  __shared__ __attribute__((aligned(64))) bf16 kt[2][32 * 64];  // [key][d]
  __shared__ __attribute__((aligned(64))) bf16 vt[2][64 * 32];  // [d][key]
  __shared__ __attribute__((aligned(64))) bf16 plds[4][16 * 32];

  int tid  = threadIdx.x;
  int lane = tid & 31;
  int wv   = tid >> 5;
  int head = blockIdx.y;           // b*16 + n
  int b = head >> 4, n = head & 15;
  int q0 = blockIdx.x * 64 + wv * 16;

  const bf16* Qh = Q  + (size_t)head * S * Dh;
  const bf16* Kh = Kd + (size_t)head * S * Dh;
  const bf16* Vh = Vt + (size_t)head * Dh * S;

  // Stage a 32-key K/V tile into LDS buffer `buf`. 16B chunks; 4 async
  // instructions per wave per tile (K tile is contiguous 4KB; V rows stride S).
  auto issue_copy = [&](int buf, int t0) {
    const bf16* gK = Kh + (size_t)t0 * Dh;
    const bf16* gV = Vh + t0;
    bf16* lK = &kt[buf][0];
    bf16* lV = &vt[buf][0];
    int c0 = tid, c1 = tid + 128;
#if HAVE_ASYNC_LDS
    __builtin_amdgcn_global_load_async_to_lds_b128(
        (AS_GLOBAL v4i*)(gK + 8 * c0), (AS_LDS v4i*)(lK + 8 * c0), 0, 0);
    __builtin_amdgcn_global_load_async_to_lds_b128(
        (AS_GLOBAL v4i*)(gK + 8 * c1), (AS_LDS v4i*)(lK + 8 * c1), 0, 0);
    __builtin_amdgcn_global_load_async_to_lds_b128(
        (AS_GLOBAL v4i*)(gV + (size_t)(c0 >> 2) * S + (c0 & 3) * 8),
        (AS_LDS v4i*)(lV + 8 * c0), 0, 0);
    __builtin_amdgcn_global_load_async_to_lds_b128(
        (AS_GLOBAL v4i*)(gV + (size_t)(c1 >> 2) * S + (c1 & 3) * 8),
        (AS_LDS v4i*)(lV + 8 * c1), 0, 0);
#else
    *(v8bf*)(lK + 8 * c0) = *(const v8bf*)(gK + 8 * c0);
    *(v8bf*)(lK + 8 * c1) = *(const v8bf*)(gK + 8 * c1);
    *(v8bf*)(lV + 8 * c0) = *(const v8bf*)(gV + (size_t)(c0 >> 2) * S + (c0 & 3) * 8);
    *(v8bf*)(lV + 8 * c1) = *(const v8bf*)(gV + (size_t)(c1 >> 2) * S + (c1 & 3) * 8);
#endif
  };

  v16bf aQ0 = load_a_frag(Qh + (size_t)q0 * Dh + 0,  Dh, lane);
  v16bf aQ1 = load_a_frag(Qh + (size_t)q0 * Dh + 32, Dh, lane);

  v8f o0, o1, o2, o3;
#pragma unroll
  for (int i = 0; i < 8; ++i) { o0[i] = 0.f; o1[i] = 0.f; o2[i] = 0.f; o3[i] = 0.f; }
  float mi[8], li[8];
#pragma unroll
  for (int r = 0; r < 8; ++r) { mi[r] = -1e30f; li[r] = 0.f; }

  int hh = lane >> 4;
  int cc = lane & 15;

  issue_copy(0, 0);

  const int T = S / 32;  // 64 key tiles
  for (int it = 0; it < T; ++it) {
    int cur = it & 1;
    bool have_next = (it + 1 < T);
    if (have_next) issue_copy(cur ^ 1, (it + 1) * 32);
#if HAVE_ASYNC_LDS
    if (have_next) __builtin_amdgcn_s_wait_asynccnt(4);
    else           __builtin_amdgcn_s_wait_asynccnt(0);
#endif
    __syncthreads();  // current tile resident for all waves

    const bf16* lK = &kt[cur][0];
    const bf16* lV = &vt[cur][0];

    // ---- S = Q @ K^T for 16 rows x 32 keys (Q pre-scaled by 1/sqrt(D)) ----
    v8f s0, s1;
#pragma unroll
    for (int i = 0; i < 8; ++i) { s0[i] = 0.f; s1[i] = 0.f; }
    v16bf kb;
    kb = load_b_frag(lK + 0 * 64 + 0,  64, lane);
    s0 = __builtin_amdgcn_wmma_f32_16x16x32_bf16(false, aQ0, false, kb, (short)0, s0, false, false);
    kb = load_b_frag(lK + 0 * 64 + 32, 64, lane);
    s0 = __builtin_amdgcn_wmma_f32_16x16x32_bf16(false, aQ1, false, kb, (short)0, s0, false, false);
    kb = load_b_frag(lK + 16 * 64 + 0,  64, lane);
    s1 = __builtin_amdgcn_wmma_f32_16x16x32_bf16(false, aQ0, false, kb, (short)0, s1, false, false);
    kb = load_b_frag(lK + 16 * 64 + 32, 64, lane);
    s1 = __builtin_amdgcn_wmma_f32_16x16x32_bf16(false, aQ1, false, kb, (short)0, s1, false, false);

    // ---- online softmax (row reductions across the 16-lane half) ----
#pragma unroll
    for (int r = 0; r < 8; ++r) {
      float t = fmaxf(s0[r], s1[r]);
      t = fmaxf(t, __shfl_xor(t, 1, 32));
      t = fmaxf(t, __shfl_xor(t, 2, 32));
      t = fmaxf(t, __shfl_xor(t, 4, 32));
      t = fmaxf(t, __shfl_xor(t, 8, 32));
      float mnew  = fmaxf(mi[r], t);
      float alpha = __expf(mi[r] - mnew);
      float p0 = __expf(s0[r] - mnew);
      float p1 = __expf(s1[r] - mnew);
      float rs = p0 + p1;
      rs += __shfl_xor(rs, 1, 32);
      rs += __shfl_xor(rs, 2, 32);
      rs += __shfl_xor(rs, 4, 32);
      rs += __shfl_xor(rs, 8, 32);
      li[r] = li[r] * alpha + rs;
      mi[r] = mnew;
      o0[r] *= alpha; o1[r] *= alpha; o2[r] *= alpha; o3[r] *= alpha;
      plds[wv][(r + 8 * hh) * 32 + cc]      = f2bf(p0);
      plds[wv][(r + 8 * hh) * 32 + 16 + cc] = f2bf(p1);
    }

    // ---- reload P as an A-fragment (same wave; DS waits compiler-managed) --
    v16bf pa;
    {
      const bf16* pp = &plds[wv][cc * 32 + (hh << 3)];
      v8bf lo = *(const v8bf*)(pp);
      v8bf hi = *(const v8bf*)(pp + 16);
#pragma unroll
      for (int i = 0; i < 8; ++i) { pa[i] = lo[i]; pa[i + 8] = hi[i]; }
    }

    // ---- O += P @ V (V tile transposed in LDS: contiguous along keys) ----
    v16bf vb;
    vb = load_b_frag(lV + 0  * 32, 32, lane);
    o0 = __builtin_amdgcn_wmma_f32_16x16x32_bf16(false, pa, false, vb, (short)0, o0, false, false);
    vb = load_b_frag(lV + 16 * 32, 32, lane);
    o1 = __builtin_amdgcn_wmma_f32_16x16x32_bf16(false, pa, false, vb, (short)0, o1, false, false);
    vb = load_b_frag(lV + 32 * 32, 32, lane);
    o2 = __builtin_amdgcn_wmma_f32_16x16x32_bf16(false, pa, false, vb, (short)0, o2, false, false);
    vb = load_b_frag(lV + 48 * 32, 32, lane);
    o3 = __builtin_amdgcn_wmma_f32_16x16x32_bf16(false, pa, false, vb, (short)0, o3, false, false);

    __syncthreads();  // all waves done reading `cur` before it is refilled
  }

  // ---- finalize: divide by row sums, store bf16 into (B,S,H) layout ----
#pragma unroll
  for (int r = 0; r < 8; ++r) {
    float inv = 1.0f / li[r];
    int row = q0 + r + 8 * hh;
    size_t base = ((size_t)(b * S + row)) * Hd + n * Dh;
    Aout[base + 0  + cc] = f2bf(o0[r] * inv);
    Aout[base + 16 + cc] = f2bf(o1[r] * inv);
    Aout[base + 32 + cc] = f2bf(o2[r] * inv);
    Aout[base + 48 + cc] = f2bf(o3[r] * inv);
  }
}

// ---------------------------------------------------------------------------
// Launcher
// ---------------------------------------------------------------------------
extern "C" void kernel_launch(void* const* d_in, const int* in_sizes, int n_in,
                              void* d_out, int out_size, void* d_ws, size_t ws_size,
                              hipStream_t stream) {
  (void)in_sizes; (void)n_in; (void)out_size; (void)ws_size;
  const float* hidden = (const float*)d_in[0];
  const float* cosb   = (const float*)d_in[1];
  const float* sinb   = (const float*)d_in[2];
  const float* q_w    = (const float*)d_in[3];
  const float* q_b    = (const float*)d_in[4];
  const float* kv_w   = (const float*)d_in[5];
  const float* kv_b   = (const float*)d_in[6];
  const float* qn_w   = (const float*)d_in[7];
  const float* qn_b   = (const float*)d_in[8];
  const float* kn_w   = (const float*)d_in[9];
  const float* kn_b   = (const float*)d_in[10];
  const float* o_w    = (const float*)d_in[11];
  const float* o_b    = (const float*)d_in[12];
  float* out = (float*)d_out;

  const int Bz = 2, S = 2048, H = 1024;
  const int M = Bz * S;  // 4096

  char* ws = (char*)d_ws;
  size_t off = 0;
  bf16*  hid_bf = (bf16*)(ws + off);  off += (size_t)M * H * 2;
  bf16*  qw_bf  = (bf16*)(ws + off);  off += (size_t)H * H * 2;
  bf16*  kvw_bf = (bf16*)(ws + off);  off += (size_t)2 * H * H * 2;
  bf16*  ow_bf  = (bf16*)(ws + off);  off += (size_t)H * H * 2;
  float* q_f    = (float*)(ws + off); off += (size_t)M * H * 4;
  float* kv_f   = (float*)(ws + off); off += (size_t)M * 2 * H * 4;
  bf16*  q_bf   = (bf16*)(ws + off);  off += (size_t)M * H * 2;
  bf16*  k_bf   = (bf16*)(ws + off);  off += (size_t)M * H * 2;
  bf16*  vT_bf  = (bf16*)(ws + off);  off += (size_t)M * H * 2;
  bf16*  at_bf  = (bf16*)(ws + off);  off += (size_t)M * H * 2;

  auto cgrid = [](int n) { return (n + 1023) / 1024; };
  cast_f32_bf16<<<cgrid(M * H),     256, 0, stream>>>(hidden, hid_bf, M * H);
  cast_f32_bf16<<<cgrid(H * H),     256, 0, stream>>>(q_w,  qw_bf,  H * H);
  cast_f32_bf16<<<cgrid(2 * H * H), 256, 0, stream>>>(kv_w, kvw_bf, 2 * H * H);
  cast_f32_bf16<<<cgrid(H * H),     256, 0, stream>>>(o_w,  ow_bf,  H * H);

  // Q projection: (4096x1024) = hidden @ q_w^T + q_b
  {
    int waves = (M / 32) * (H / 64);
    gemm_bf16_kernel<<<waves / 8, 256, 0, stream>>>(hid_bf, qw_bf, q_b,
                                                    q_f, M, H, H);
  }
  // KV projection: (4096x2048) = hidden @ kv_w^T + kv_b
  {
    int waves = (M / 32) * ((2 * H) / 64);
    gemm_bf16_kernel<<<waves / 8, 256, 0, stream>>>(hid_bf, kvw_bf, kv_b,
                                                    kv_f, M, 2 * H, H);
  }
  // LN + RoPE + layout transform (Q scaled by 1/sqrt(D); V transposed)
  ln_rope_kernel<<<(M * 16) / 8, 256, 0, stream>>>(
      q_f, kv_f, cosb, sinb, qn_w, qn_b, kn_w, kn_b, q_bf, k_bf, vT_bf);

  // Flash attention
  flash_attn_kernel<<<dim3(S / 64, Bz * 16), 128, 0, stream>>>(
      q_bf, k_bf, vT_bf, at_bf);

  // Output projection -> fp32 d_out (attn output is bf16 in ws)
  {
    int waves = (M / 32) * (H / 64);
    gemm_bf16_kernel<<<waves / 8, 256, 0, stream>>>(at_bf, ow_bf, o_b,
                                                    out, M, H, H);
  }
}
